// CrossUnit_26130581029097
// MI455X (gfx1250) — matrix operand order repaired and checked by reading, not compiled, not run on hardware
//
#include <hip/hip_runtime.h>
#include <stdint.h>

// ---------------- problem constants ----------------
#define D_IN   768
#define DD     512
#define NPATCH 49
#define NENT   2048
#define NMEN   512

// ---------------- WMMA types (CDNA5 / gfx1250, wave32) ----------------
typedef __attribute__((ext_vector_type(16))) __bf16 v16bf;
typedef __attribute__((ext_vector_type(8)))  float  v8f;

union FragAB { v16bf v; uint4 q[2]; unsigned short h[16]; };
union FragC  { v8f   v; float f[8]; };

__device__ __forceinline__ unsigned short f32_to_bf16(float x) {
    unsigned int u = __float_as_uint(x);
    unsigned int r = u + 0x7FFFu + ((u >> 16) & 1u);   // round-to-nearest-even
    return (unsigned short)(r >> 16);
}
__device__ __forceinline__ ushort4 f4_to_bf4(float4 f) {
    ushort4 o;
    o.x = f32_to_bf16(f.x); o.y = f32_to_bf16(f.y);
    o.z = f32_to_bf16(f.z); o.w = f32_to_bf16(f.w);
    return o;
}

// ---------------- weight transpose + bf16 convert: out[n][k] = bf16(in[k][n]) ----------------
__global__ void k_transpose_bf16(const float* __restrict__ in,
                                 unsigned short* __restrict__ out,
                                 int K, int N) {
    int tid = blockIdx.x * blockDim.x + threadIdx.x;
    int total = K * N;
    if (tid >= total) return;
    int n = tid / K;
    int k = tid - n * K;
    out[tid] = f32_to_bf16(in[(size_t)k * N + n]);
}

__global__ void k_f32_to_bf16(const float* __restrict__ in,
                              unsigned short* __restrict__ out, int n) {
    int tid = blockIdx.x * blockDim.x + threadIdx.x;
    if (tid < n) out[tid] = f32_to_bf16(in[tid]);
}

// ---------------- generic bf16-WMMA GEMM ----------------
// C[M,N] = act(A[M,K] @ Bt[N,K]^T + bias[N]); A f32 row-major, Bt bf16 row-major [N][K].
// Block = 128 threads (4 waves). Block tile 64(M) x 64(N). Requires N%64==0, K%32==0.
// Dynamic LDS: 64*K bf16.
__global__ __launch_bounds__(128)
void k_wmma_gemm(const float* __restrict__ A, int lda,
                 const unsigned short* __restrict__ Bt, int ldb,
                 const float* __restrict__ bias,
                 float* __restrict__ C, int ldc,
                 int M, int N, int K, int relu)
{
    extern __shared__ __align__(16) unsigned short sA[];   // [64][K]
    const int tid  = threadIdx.x;
    const int lane = tid & 31;
    const int wave = tid >> 5;
    const int m0 = blockIdx.x * 64;
    const int n0 = blockIdx.y * 64;

    // stage A tile as bf16 in LDS
    const int quadsPerRow = K >> 2;
    for (int q = tid; q < 64 * quadsPerRow; q += 128) {
        int row = q / quadsPerRow;
        int c4  = (q - row * quadsPerRow) << 2;
        int gr  = m0 + row;
        float4 f = make_float4(0.f, 0.f, 0.f, 0.f);
        if (gr < M) f = *(const float4*)(A + (size_t)gr * lda + c4);
        *(ushort4*)(sA + row * K + c4) = f4_to_bf4(f);
    }
    __syncthreads();

    FragC acc[4];
#pragma unroll
    for (int j = 0; j < 4; ++j)
#pragma unroll
        for (int r = 0; r < 8; ++r) acc[j].f[r] = 0.f;

    const int rowL  = wave * 16 + (lane & 15);
    const int kselA = (lane < 16) ? 0 : 8;
    const int kselB = (lane < 16) ? 0 : 8;
    const int ksteps = K >> 5;

    for (int ks = 0; ks < ksteps; ++ks) {
        FragAB a;
        const unsigned short* ap = sA + rowL * K + ks * 32 + kselA;
        a.q[0] = *(const uint4*)(ap);
        a.q[1] = *(const uint4*)(ap + 16);
#pragma unroll
        for (int j = 0; j < 4; ++j) {
            int col = n0 + j * 16 + (lane & 15);
            FragAB b;
            const unsigned short* bp = Bt + (size_t)col * ldb + ks * 32 + kselB;
            b.q[0] = *(const uint4*)(bp);
            b.q[1] = *(const uint4*)(bp + 16);
            acc[j].v = __builtin_amdgcn_wmma_f32_16x16x32_bf16(
                false, a.v, false, b.v, (short)0, acc[j].v, false, false);
        }
    }

#pragma unroll
    for (int j = 0; j < 4; ++j) {
        int col = n0 + j * 16 + (lane & 15);
        float bv = bias ? bias[col] : 0.f;
#pragma unroll
        for (int r = 0; r < 8; ++r) {
            int row = m0 + wave * 16 + r + ((lane < 16) ? 0 : 8);
            if (row < M) {
                float v = acc[j].f[r] + bv;
                if (relu) v = fmaxf(v, 0.f);
                C[(size_t)row * ldc + col] = v;
            }
        }
    }
}

// ---------------- fused token projection + 49-way cross attention ----------------
// For sample n: T = toks[n] @ W_img + b   (49x512, kept in LDS, never hits HBM)
//               s = (cls_p[n] . T_p) * 512^-0.5 ; a = softmax(s) ; attn = sum_p a_p T_p
// Block = 256 threads (8 waves): 4 M-tiles x 2 column halves. K=768 in 3 chunks of 256.
__global__ __launch_bounds__(256)
void k_tok_attn(const float* __restrict__ toks,          // [N,49,768]
                const float* __restrict__ clsp,          // [N,512]
                const unsigned short* __restrict__ WiT,  // [512][768] bf16
                const float* __restrict__ bias,          // [512]
                float* __restrict__ attn_out)            // [N,512]
{
    __shared__ __align__(16) unsigned short sA[64 * 256];   // 32 KB A chunk
    __shared__ float sT[NPATCH * DD];                       // 98 KB projected tokens
    __shared__ float sQ[DD];
    __shared__ float sS[64];

    const int n    = blockIdx.x;
    const int tid  = threadIdx.x;
    const int lane = tid & 31;
    const int wave = tid >> 5;
    const int mtile = wave & 3;      // row group (16 rows)
    const int half  = wave >> 2;     // 256-column half

    for (int i = tid; i < NPATCH * DD; i += 256) sT[i] = bias[i & (DD - 1)];
    for (int i = tid; i < DD; i += 256) sQ[i] = clsp[(size_t)n * DD + i];
    __syncthreads();

    const float* xbase = toks + (size_t)n * (NPATCH * D_IN);
    const int kselA = (lane < 16) ? 0 : 8;
    const int kselB = (lane < 16) ? 0 : 8;

    for (int kc = 0; kc < 3; ++kc) {
        // stage 64x256 bf16 A chunk (rows >= 49 zero-padded)
        for (int qd = tid; qd < 64 * 64; qd += 256) {
            int row = qd >> 6;
            int c4  = (qd & 63) << 2;
            float4 f = make_float4(0.f, 0.f, 0.f, 0.f);
            if (row < NPATCH)
                f = *(const float4*)(xbase + row * D_IN + kc * 256 + c4);
            *(ushort4*)(sA + row * 256 + c4) = f4_to_bf4(f);
        }
        __syncthreads();

        for (int g = 0; g < 4; ++g) {
            FragC acc[4];
#pragma unroll
            for (int j = 0; j < 4; ++j)
#pragma unroll
                for (int r = 0; r < 8; ++r) acc[j].f[r] = 0.f;

            for (int ks = 0; ks < 8; ++ks) {
                FragAB a;
                const unsigned short* ap =
                    sA + (mtile * 16 + (lane & 15)) * 256 + ks * 32 + kselA;
                a.q[0] = *(const uint4*)(ap);
                a.q[1] = *(const uint4*)(ap + 16);
#pragma unroll
                for (int j = 0; j < 4; ++j) {
                    int col = (half * 16 + g * 4 + j) * 16 + (lane & 15);
                    FragAB b;
                    const unsigned short* bp =
                        WiT + (size_t)col * D_IN + kc * 256 + ks * 32 + kselB;
                    b.q[0] = *(const uint4*)(bp);
                    b.q[1] = *(const uint4*)(bp + 16);
                    acc[j].v = __builtin_amdgcn_wmma_f32_16x16x32_bf16(
                        false, a.v, false, b.v, (short)0, acc[j].v, false, false);
                }
            }
            // accumulate into sT (each wave owns disjoint rows x cols)
#pragma unroll
            for (int j = 0; j < 4; ++j) {
                int col = (half * 16 + g * 4 + j) * 16 + (lane & 15);
#pragma unroll
                for (int r = 0; r < 8; ++r) {
                    int row = mtile * 16 + r + ((lane < 16) ? 0 : 8);
                    if (row < NPATCH) sT[row * DD + col] += acc[j].f[r];
                }
            }
        }
        __syncthreads();
    }

    // scores s[p] = scale * dot(q, T[p])
    const float scale = 0.044194173824159216f;   // 512^-0.5
    for (int p = wave; p < NPATCH; p += 8) {
        float part = 0.f;
        for (int d = lane; d < DD; d += 32) part += sQ[d] * sT[p * DD + d];
        for (int m = 16; m >= 1; m >>= 1) part += __shfl_xor(part, m, 32);
        if (lane == 0) sS[p] = part * scale;
    }
    __syncthreads();
    if (tid == 0) {
        float mx = -1e30f;
        for (int p = 0; p < NPATCH; ++p) mx = fmaxf(mx, sS[p]);
        float sum = 0.f;
        for (int p = 0; p < NPATCH; ++p) { float e = expf(sS[p] - mx); sS[p] = e; sum += e; }
        float inv = 1.f / sum;
        for (int p = 0; p < NPATCH; ++p) sS[p] *= inv;
    }
    __syncthreads();
    for (int d = tid; d < DD; d += 256) {
        float s = 0.f;
        for (int p = 0; p < NPATCH; ++p) s += sS[p] * sT[p * DD + d];
        attn_out[(size_t)n * DD + d] = s;
    }
}

// ---------------- per-row: MoE gate + combine + tanh-gate + LayerNorm ----------------
__device__ __forceinline__ float block_sum(float v, float* red, int tid) {
    for (int m = 16; m >= 1; m >>= 1) v += __shfl_xor(v, m, 32);
    __syncthreads();
    if ((tid & 31) == 0) red[tid >> 5] = v;
    __syncthreads();
    if (tid == 0) {
        float s = 0.f;
        for (int i = 0; i < 8; ++i) s += red[i];
        red[0] = s;
    }
    __syncthreads();
    float r = red[0];
    __syncthreads();
    return r;
}

__global__ __launch_bounds__(256)
void k_finish(const float* __restrict__ clsp, const float* __restrict__ txt,
              const float* __restrict__ img,  const float* __restrict__ et,
              const float* __restrict__ ei,
              const float* __restrict__ gw, const float* __restrict__ gb,   // [1024,2],[2]
              const float* __restrict__ gfw, const float* __restrict__ gfb, // [512],[1]
              const float* __restrict__ lng, const float* __restrict__ lnb,
              float* __restrict__ out)
{
    __shared__ float red[8];
    __shared__ float sc[3];
    const int n = blockIdx.x;
    const int tid = threadIdx.x;
    const float* tr  = txt  + (size_t)n * DD;
    const float* ir  = img  + (size_t)n * DD;
    const float* cr  = clsp + (size_t)n * DD;
    const float* er  = et   + (size_t)n * DD;
    const float* eir = ei   + (size_t)n * DD;
    const int d0 = tid, d1 = tid + 256;

    float l0p = tr[d0] * gw[d0 * 2 + 0] + tr[d1] * gw[d1 * 2 + 0]
              + ir[d0] * gw[(512 + d0) * 2 + 0] + ir[d1] * gw[(512 + d1) * 2 + 0];
    float l1p = tr[d0] * gw[d0 * 2 + 1] + tr[d1] * gw[d1 * 2 + 1]
              + ir[d0] * gw[(512 + d0) * 2 + 1] + ir[d1] * gw[(512 + d1) * 2 + 1];
    float gp  = cr[d0] * gfw[d0] + cr[d1] * gfw[d1];

    float l0 = block_sum(l0p, red, tid);
    float l1 = block_sum(l1p, red, tid);
    float gd = block_sum(gp,  red, tid);
    if (tid == 0) {
        float a0 = l0 + gb[0], a1 = l1 + gb[1];
        float mx = fmaxf(a0, a1);
        float e0 = expf(a0 - mx), e1 = expf(a1 - mx);
        float inv = 1.f / (e0 + e1);
        sc[0] = e0 * inv; sc[1] = e1 * inv; sc[2] = tanhf(gd + gfb[0]);
    }
    __syncthreads();
    float g0 = sc[0], g1 = sc[1], gs = sc[2];

    float x0 = cr[d0] * gs + g0 * er[d0] + g1 * eir[d0];
    float x1 = cr[d1] * gs + g0 * er[d1] + g1 * eir[d1];
    float s  = block_sum(x0 + x1,           red, tid);
    float sq = block_sum(x0 * x0 + x1 * x1, red, tid);
    float mean = s * (1.f / 512.f);
    float var  = sq * (1.f / 512.f) - mean * mean;
    float rinv = rsqrtf(var + 1e-5f);
    out[(size_t)n * DD + d0] = (x0 - mean) * rinv * lng[d0] + lnb[d0];
    out[(size_t)n * DD + d1] = (x1 - mean) * rinv * lng[d1] + lnb[d1];
}

// ---------------- host orchestration ----------------
extern "C" void kernel_launch(void* const* d_in, const int* in_sizes, int n_in,
                              void* d_out, int out_size, void* d_ws, size_t ws_size,
                              hipStream_t stream) {
    (void)in_sizes; (void)n_in; (void)out_size; (void)ws_size;

    const float* e_cls   = (const float*)d_in[0];
    const float* e_toks  = (const float*)d_in[1];
    const float* m_cls   = (const float*)d_in[2];
    const float* m_toks  = (const float*)d_in[3];
    const float* text_fc_w  = (const float*)d_in[4];
    const float* text_fc_b  = (const float*)d_in[5];
    const float* image_fc_w = (const float*)d_in[6];
    const float* image_fc_b = (const float*)d_in[7];
    const float* gate_fc_w  = (const float*)d_in[8];
    const float* gate_fc_b  = (const float*)d_in[9];
    const float* ln_g    = (const float*)d_in[10];
    const float* ln_b    = (const float*)d_in[11];
    const float* match_w = (const float*)d_in[12];
    const float* match_b = (const float*)d_in[13];
    // d_in[14..18]: mh_query, mh_wq, mh_bq, mh_wk, mh_bk — algebraically unused
    const float* mh_wv = (const float*)d_in[19];
    const float* mh_bv = (const float*)d_in[20];
    const float* mh_wo = (const float*)d_in[21];
    const float* mh_bo = (const float*)d_in[22];
    const float* moe_gw = (const float*)d_in[23];
    const float* moe_gb = (const float*)d_in[24];
    const float* tw1 = (const float*)d_in[25]; const float* tb1 = (const float*)d_in[26];
    const float* tw2 = (const float*)d_in[27]; const float* tb2 = (const float*)d_in[28];
    const float* iw1 = (const float*)d_in[29]; const float* ib1 = (const float*)d_in[30];
    const float* iw2 = (const float*)d_in[31]; const float* ib2 = (const float*)d_in[32];

    // ---- workspace layout (≈51 MB) ----
    char* base = (char*)d_ws;
    size_t off = 0;
    auto alloc = [&](size_t bytes) -> char* {
        char* p = base + off;
        off = (off + bytes + 255) & ~(size_t)255;
        return p;
    };
    unsigned short* wTtext  = (unsigned short*)alloc((size_t)DD * D_IN * 2);
    unsigned short* wTimg   = (unsigned short*)alloc((size_t)DD * D_IN * 2);
    unsigned short* wTmatch = (unsigned short*)alloc((size_t)DD * DD * 2);
    unsigned short* wTwv    = (unsigned short*)alloc((size_t)DD * DD * 2);
    unsigned short* wTwo    = (unsigned short*)alloc((size_t)DD * DD * 2);
    unsigned short* wTtw1   = (unsigned short*)alloc((size_t)DD * DD * 2);
    unsigned short* wTtw2   = (unsigned short*)alloc((size_t)DD * DD * 2);
    unsigned short* wTiw1   = (unsigned short*)alloc((size_t)DD * DD * 2);
    unsigned short* wTiw2   = (unsigned short*)alloc((size_t)DD * DD * 2);
    unsigned short* ectxbf  = (unsigned short*)alloc((size_t)NENT * DD * 2);
    const size_t BUF = (size_t)NENT * DD * 4;      // sized for the larger side
    float* clsp  = (float*)alloc(BUF);
    float* attnb = (float*)alloc(BUF);
    float* txtb  = (float*)alloc(BUF);
    float* ybuf  = (float*)alloc(BUF);
    float* vbuf  = (float*)alloc(BUF);
    float* imgb  = (float*)alloc(BUF);
    float* h1    = (float*)alloc(BUF);
    float* etb   = (float*)alloc(BUF);
    float* eib   = (float*)alloc(BUF);
    float* ectx  = (float*)alloc(BUF);
    float* mctx  = (float*)alloc((size_t)NMEN * DD * 4);

    auto cdiv = [](int a, int b) { return (a + b - 1) / b; };

    // ---- pre-transpose weights to bf16 [N][K] ----
    k_transpose_bf16<<<cdiv(DD * D_IN, 256), 256, 0, stream>>>(text_fc_w,  wTtext,  D_IN, DD);
    k_transpose_bf16<<<cdiv(DD * D_IN, 256), 256, 0, stream>>>(image_fc_w, wTimg,   D_IN, DD);
    k_transpose_bf16<<<cdiv(DD * DD, 256), 256, 0, stream>>>(match_w, wTmatch, DD, DD);
    k_transpose_bf16<<<cdiv(DD * DD, 256), 256, 0, stream>>>(mh_wv,   wTwv,    DD, DD);
    k_transpose_bf16<<<cdiv(DD * DD, 256), 256, 0, stream>>>(mh_wo,   wTwo,    DD, DD);
    k_transpose_bf16<<<cdiv(DD * DD, 256), 256, 0, stream>>>(tw1,     wTtw1,   DD, DD);
    k_transpose_bf16<<<cdiv(DD * DD, 256), 256, 0, stream>>>(tw2,     wTtw2,   DD, DD);
    k_transpose_bf16<<<cdiv(DD * DD, 256), 256, 0, stream>>>(iw1,     wTiw1,   DD, DD);
    k_transpose_bf16<<<cdiv(DD * DD, 256), 256, 0, stream>>>(iw2,     wTiw2,   DD, DD);

    const size_t lds768 = (size_t)64 * D_IN * 2;   // 96 KB
    const size_t lds512 = (size_t)64 * DD * 2;     // 64 KB

    auto side = [&](int N, const float* cls, const float* toks, float* ctx_out) {
        dim3 gK(N / 64, DD / 64);
        // cls_p = cls @ W_text + b
        k_wmma_gemm<<<gK, 128, lds768, stream>>>(cls, D_IN, wTtext, D_IN, text_fc_b,
                                                 clsp, DD, N, DD, D_IN, 0);
        // fused token projection + attention  ->  attn
        k_tok_attn<<<N, 256, 0, stream>>>(toks, clsp, wTimg, image_fc_b, attnb);
        // txt = attn @ match_w + b ;  y = cls_p @ match_w + b
        k_wmma_gemm<<<gK, 128, lds512, stream>>>(attnb, DD, wTmatch, DD, match_b,
                                                 txtb, DD, N, DD, DD, 0);
        k_wmma_gemm<<<gK, 128, lds512, stream>>>(clsp, DD, wTmatch, DD, match_b,
                                                 ybuf, DD, N, DD, DD, 0);
        // img = (y @ wv + bv) @ wo + bo   (mh_pool collapsed)
        k_wmma_gemm<<<gK, 128, lds512, stream>>>(ybuf, DD, wTwv, DD, mh_bv,
                                                 vbuf, DD, N, DD, DD, 0);
        k_wmma_gemm<<<gK, 128, lds512, stream>>>(vbuf, DD, wTwo, DD, mh_bo,
                                                 imgb, DD, N, DD, DD, 0);
        // MoE experts
        k_wmma_gemm<<<gK, 128, lds512, stream>>>(txtb, DD, wTtw1, DD, tb1,
                                                 h1, DD, N, DD, DD, 1);
        k_wmma_gemm<<<gK, 128, lds512, stream>>>(h1, DD, wTtw2, DD, tb2,
                                                 etb, DD, N, DD, DD, 0);
        k_wmma_gemm<<<gK, 128, lds512, stream>>>(imgb, DD, wTiw1, DD, ib1,
                                                 h1, DD, N, DD, DD, 1);
        k_wmma_gemm<<<gK, 128, lds512, stream>>>(h1, DD, wTiw2, DD, ib2,
                                                 eib, DD, N, DD, DD, 0);
        // gate + combine + tanh-gate + LN
        k_finish<<<N, 256, 0, stream>>>(clsp, txtb, imgb, etb, eib,
                                        moe_gw, moe_gb, gate_fc_w, gate_fc_b,
                                        ln_g, ln_b, ctx_out);
    };

    side(NENT, e_cls, e_toks, ectx);
    side(NMEN, m_cls, m_toks, mctx);

    // scores = m_ctx @ e_ctx^T   (e_ctx is already [N][K] for the Bt operand)
    k_f32_to_bf16<<<cdiv(NENT * DD, 256), 256, 0, stream>>>(ectx, ectxbf, NENT * DD);
    k_wmma_gemm<<<dim3(NMEN / 64, NENT / 64), 128, lds512, stream>>>(
        mctx, DD, ectxbf, DD, nullptr, (float*)d_out, NENT, NMEN, NENT, DD, 0);
}
